// Network_88734024336045
// MI455X (gfx1250) — compile-verified, matrix operand order checked by reading, not run on hardware
//
#include <hip/hip_runtime.h>

// Problem constants (match reference)
#define BB 2048
#define TT 500
#define NN 64
#define BT (BB * TT)                 // 1,024,000 rows of 64 floats
#define TILES (BT / 16)              // 64,000 tiles of 16 rows
#define RED_BLOCK_THREADS 128
#define RED_WAVES_PER_BLOCK 4
#define RED_BLOCKS 2000
#define RED_WAVES (RED_BLOCKS * RED_WAVES_PER_BLOCK) // 8,000 waves (wave32)
#define TILES_PER_WAVE (TILES / RED_WAVES)           // 8
#define LDS_ROW_F 68                 // 64 + 4 pad floats -> conflict-light ds_load
#define LDS_TILE_F (16 * LDS_ROW_F)  // 1088 floats = 4352 B per tile buffer
#define DT_STEP 0.25f

typedef __attribute__((ext_vector_type(2)))  float    v2f;
typedef __attribute__((ext_vector_type(8)))  float    v8f;
typedef __attribute__((ext_vector_type(16))) _Float16 v16h;
typedef __attribute__((ext_vector_type(4)))  int      v4i;

#if defined(__HIP_DEVICE_COMPILE__)
# if __has_builtin(__builtin_amdgcn_wmma_f32_16x16x4_f32)
#  define ATHENA_WMMA_F32 1
#  warning "ATHENA: device path = wmma_f32_16x16x4_f32"
# elif __has_builtin(__builtin_amdgcn_wmma_f32_16x16x32_f16)
#  define ATHENA_WMMA_F16 1
#  warning "ATHENA: device path = wmma_f32_16x16x32_f16 fallback"
# else
#  warning "ATHENA: device path = shuffle fallback (NO WMMA)"
# endif
# if __has_builtin(__builtin_amdgcn_global_load_async_to_lds_b128)
#  define ATHENA_ASYNC 1
#  warning "ATHENA: using global_load_async_to_lds_b128 staging"
# else
#  warning "ATHENA: async-to-LDS builtin absent; direct global loads"
# endif
#endif

// ---------------------------------------------------------------------------
// Async staging helpers (device-only; bodies compiled out elsewhere)
// ---------------------------------------------------------------------------
__device__ __forceinline__ void issue_tile_async(const float* gsrc, float* lbuf,
                                                 int m, int h) {
#if defined(ATHENA_ASYNC)
  // Each call: 32 lanes x 16B = 512B fully-contiguous global -> padded LDS rows.
  // Lane L copies row r = 2j+h, bytes [m*16, m*16+16).
#pragma unroll
  for (int j = 0; j < 8; ++j) {
    const int r = 2 * j + h;
    __builtin_amdgcn_global_load_async_to_lds_b128(
        (v4i*)(gsrc + (size_t)r * NN + (size_t)m * 4),
        (v4i*)(lbuf + (size_t)r * LDS_ROW_F + (size_t)m * 4),
        0, 0);
  }
#else
  (void)gsrc; (void)lbuf; (void)m; (void)h;
#endif
}

__device__ __forceinline__ void wait_async_le8() {
#if defined(ATHENA_ASYNC)
# if __has_builtin(__builtin_amdgcn_s_wait_asynccnt)
  __builtin_amdgcn_s_wait_asynccnt(8);
# else
  asm volatile("s_wait_asynccnt 0x8" ::: "memory");
# endif
#endif
}

__device__ __forceinline__ void wait_async_le0() {
#if defined(ATHENA_ASYNC)
# if __has_builtin(__builtin_amdgcn_s_wait_asynccnt)
  __builtin_amdgcn_s_wait_asynccnt(0);
# else
  asm volatile("s_wait_asynccnt 0x0" ::: "memory");
# endif
#endif
}

// ---------------------------------------------------------------------------
// Reduce 8 float4s to (p0, p1): sums of first/second 16 floats of a half-row.
// ---------------------------------------------------------------------------
__device__ __forceinline__ void reduce_half_row(const float4* p, float& p0, float& p1) {
  const float4 x0 = p[0], x1 = p[1], x2 = p[2], x3 = p[3];
  const float4 x4 = p[4], x5 = p[5], x6 = p[6], x7 = p[7];
  p0 = ((x0.x + x0.y) + (x0.z + x0.w)) + ((x1.x + x1.y) + (x1.z + x1.w)) +
       ((x2.x + x2.y) + (x2.z + x2.w)) + ((x3.x + x3.y) + (x3.z + x3.w));
  p1 = ((x4.x + x4.y) + (x4.z + x4.w)) + ((x5.x + x5.y) + (x5.z + x5.w)) +
       ((x6.x + x6.y) + (x6.z + x6.w)) + ((x7.x + x7.y) + (x7.z + x7.w));
}

// WMMA row-sum finish + store: (p0,p1) sit exactly in the f32 16x4 A-fragment
// (lanes 0-15 hold K0,K1; lanes 16-31 hold K2,K3). All-ones B => D = row sums.
__device__ __forceinline__ void wmma_finish_store(float p0, float p1,
                                                  float* S, size_t r0,
                                                  int m, int h) {
#if defined(ATHENA_WMMA_F32)
  v2f a = {p0, p1};
  v2f bones = {1.0f, 1.0f};
  v8f c = {};
  c = __builtin_amdgcn_wmma_f32_16x16x4_f32(false, a, false, bones,
                                            (short)0, c, false, false);
  if (m == 0) {
    // lane 0 holds rows 0-7 (N=0); lane 16 holds rows 8-15 (N=0)
    float* sp = S + r0 + (size_t)h * 8;
    ((float4*)sp)[0] = make_float4(c[0], c[1], c[2], c[3]);
    ((float4*)sp)[1] = make_float4(c[4], c[5], c[6], c[7]);
  }
#elif defined(ATHENA_WMMA_F16)
  v16h a = {};
  a[0] = (_Float16)p0;
  a[1] = (_Float16)p1;
  v16h bones = {1, 1, 1, 1, 1, 1, 1, 1, 1, 1, 1, 1, 1, 1, 1, 1};
  v8f c = {};
  c = __builtin_amdgcn_wmma_f32_16x16x32_f16(false, a, false, bones,
                                             (short)0, c, false, false);
  if (m == 0) {
    float* sp = S + r0 + (size_t)h * 8;
    ((float4*)sp)[0] = make_float4(c[0], c[1], c[2], c[3]);
    ((float4*)sp)[1] = make_float4(c[4], c[5], c[6], c[7]);
  }
#else
  float tot = p0 + p1;
  tot += __shfl_down(tot, 16, 32);
  if (h == 0) S[r0 + m] = tot;
#endif
}

// ---------------------------------------------------------------------------
// Kernel 1: row sums S[r] = sum_n input[r*64 + n], 16 rows per wave-tile.
// ---------------------------------------------------------------------------
__global__ __launch_bounds__(RED_BLOCK_THREADS) void rowsum_wmma(
    const float* __restrict__ in, float* __restrict__ S) {
  const int wib  = (int)threadIdx.x >> 5;
  const int wave = blockIdx.x * RED_WAVES_PER_BLOCK + wib;
  const int lane = (int)threadIdx.x & 31;
  const int m = lane & 15;   // row within tile
  const int h = lane >> 4;   // half of the 64-wide row

#if defined(ATHENA_ASYNC)
  __shared__ float lds[RED_WAVES_PER_BLOCK][2][LDS_TILE_F];

  // Prologue: stage tile 0 into buffer 0.
  issue_tile_async(in + (size_t)wave * 16 * NN, &lds[wib][0][0], m, h);

#pragma unroll
  for (int i = 0; i < TILES_PER_WAVE; ++i) {
    const int tile = wave + i * RED_WAVES;
    const size_t r0 = (size_t)tile * 16;

    if (i + 1 < TILES_PER_WAVE) {
      // Double-buffer: stage tile i+1 before consuming tile i.
      const size_t g1 = ((size_t)tile + RED_WAVES) * 16 * NN;
      issue_tile_async(in + g1, &lds[wib][(i + 1) & 1][0], m, h);
      wait_async_le8();   // first 8 async ops (tile i) retired, in order
    } else {
      wait_async_le0();
    }

    const float* buf = &lds[wib][i & 1][0];
    const float4* p = (const float4*)(buf + (size_t)m * LDS_ROW_F + (size_t)h * 32);
    float p0, p1;
    reduce_half_row(p, p0, p1);
    wmma_finish_store(p0, p1, S, r0, m, h);
  }
#else
  const size_t laneoff = (size_t)m * NN + (size_t)h * 32;
  for (int i = 0; i < TILES_PER_WAVE; ++i) {
    const int tile = wave + i * RED_WAVES;
    const size_t r0 = (size_t)tile * 16;
    const float4* p = (const float4*)(in + r0 * NN + laneoff);
    if (i + 1 < TILES_PER_WAVE) {
      __builtin_prefetch((const void*)(in + (r0 + (size_t)RED_WAVES * 16) * NN + laneoff), 0, 0);
    }
    float p0, p1;
    reduce_half_row(p, p0, p1);
    wmma_finish_store(p0, p1, S, r0, m, h);
  }
#endif
}

// ---------------------------------------------------------------------------
// Kernel 2: sequential Izhikevich scan. One thread per (batch, channel);
// the two channels are fully independent given S.
// ---------------------------------------------------------------------------
__device__ __forceinline__ float izh(float& v, float& u, float I,
                                     float a, float b, float c, float d) {
  const float vn = v + DT_STEP * (0.04f * v * v + 5.0f * v + 140.0f - u + I);
  const float un = u + DT_STEP * a * (b * v - u);
  const float z = (vn >= 30.0f) ? 1.0f : 0.0f;
  v = (z > 0.0f) ? c : vn;
  u = un + z * d;
  return z;
}

__global__ __launch_bounds__(256) void izh_scan(const float* __restrict__ S,
                                                const float* __restrict__ w,
                                                float* __restrict__ out) {
  const int tid = blockIdx.x * blockDim.x + threadIdx.x; // 0 .. 2*BB-1
  const bool ch1 = tid < BB;                             // wave-uniform split
  const int b = ch1 ? tid : tid - BB;

  // shared weights
  const float w2 = w[2], w3 = w[3], w9 = w[9], w10 = w[10];
  // per-channel weights
  const float wzp = ch1 ? w[0]  : w[12];
  const float wi  = ch1 ? w[1]  : w[13];
  const float wE0 = ch1 ? w[4]  : w[16];
  const float wE1 = ch1 ? w[5]  : w[17];
  const float wIf = ch1 ? w[6]  : w[18];
  const float wTn = ch1 ? w[8]  : w[20];
  const float wMn = ch1 ? w[11] : w[23];

  // initial states: v0 and u0 = b*v0 per neuron type
  float vL = -70.0f, uL = -14.0f;  // LLBN (b=0.20)
  float vE = -64.0f, uE = -16.0f;  // EBN  (b=0.25)
  float vI = -64.0f, uI = -16.0f;  // IFN  (b=0.25)
  float vT = -70.0f, uT = -14.0f;  // TN   (b=0.20)
  float vM = -64.0f, uM = -16.0f;  // MN = IFN
  float zL = 0.0f, zT = 0.0f;

  float* o0 = out;                     // o_spikes   [B,T]
  float* o1 = out + (size_t)BT;        // v          [B,T]
  float* o2 = out + (size_t)2 * BT;    // o_spikes2  [B,T]
  float* o3 = out + (size_t)3 * BT;    // v2         [B,T]
  float* o4 = out + (size_t)4 * BT;    // o_spikes_o [B,T,4]
  float* o5 = out + (size_t)8 * BT;    // v_o        [B,T,4]

  const float* Sb = S + (size_t)b * TT;

  for (int t = 0; t < TT; ++t) {
    const float zp = wzp * Sb[t];

    const float z2 = izh(vL, uL, w2 * (zp * wi) + w3 * zL,
                         0.02f, 0.20f, -65.0f, 6.0f);
    zL = z2;
    const float z3 = izh(vE, uE, zp * wE0 + z2 * wE1,
                         0.02f, 0.25f, -55.0f, 0.05f);
    const float z4 = izh(vI, uI, z3 * wIf,
                         0.02f, 0.25f, -65.0f, 6.0f);
    const float z5 = izh(vT, uT, w9 * (z3 * wTn) + w10 * zT,
                         0.02f, 0.20f, -50.0f, 2.0f);
    zT = z5;
    const float z6 = izh(vM, uM, z5 * wMn,
                         0.02f, 0.25f, -65.0f, 6.0f);

    const size_t idx = (size_t)b * TT + t;
    if (ch1) {
      o0[idx] = z6;
      o1[idx] = vM;
      ((float4*)o4)[idx] = make_float4(z2, z3, z4, z5);
      ((float4*)o5)[idx] = make_float4(vL, vE, vI, vT);
    } else {
      o2[idx] = z6;
      o3[idx] = vM;
    }
  }
}

// ---------------------------------------------------------------------------
extern "C" void kernel_launch(void* const* d_in, const int* in_sizes, int n_in,
                              void* d_out, int out_size, void* d_ws, size_t ws_size,
                              hipStream_t stream) {
  (void)in_sizes; (void)n_in; (void)out_size; (void)ws_size;
  const float* in = (const float*)d_in[0];   // [B,T,N] f32
  const float* w  = (const float*)d_in[1];   // [24]    f32
  float* out = (float*)d_out;                // 12*B*T  f32
  float* S   = (float*)d_ws;                 // B*T row sums (4 MB scratch)

  rowsum_wmma<<<dim3(RED_BLOCKS), dim3(RED_BLOCK_THREADS), 0, stream>>>(in, S);
  izh_scan<<<dim3((2 * BB) / 256), dim3(256), 0, stream>>>(S, w, out);
}
